// CloudCrop_47064251629731
// MI455X (gfx1250) — matrix operand order repaired and checked by reading, not compiled, not run on hardware
//
#include <hip/hip_runtime.h>
#include <hip/hip_bf16.h>

typedef __bf16 bf16;
typedef __attribute__((ext_vector_type(16))) __bf16 v16bf;
typedef __attribute__((ext_vector_type(8)))  __bf16 v8bf;
typedef __attribute__((ext_vector_type(8)))  float  v8f;

#define NPTS     1024
#define CFEAT    512
#define NS       64
#define K1_REAL  515
#define K1       544          // padded to multiple of 32
#define GSTRIDE  552          // LDS row stride for grouped (bank skew)
#define H1STRIDE 264          // LDS row stride for h1 (bank skew)

// ---- workspace layout (bytes), all 256-aligned ----
#define WS_W1BF   0                      // 256*544*2 = 278528
#define WS_W2BF   278528                 // 256*256*2 = 131072
#define WS_SCALE1 409600                 // 256*4
#define WS_BIAS1  410624
#define WS_SCALE2 411648
#define WS_BIAS2  412672
#define WS_SELIDX 413696                 // 2048*64*4 = 524288
#define WS_SELLOC 937984                 // 2048*64*3*4 = 1572864  (end ~2.4MB)

// ---------------- prep: fold BN, convert weights to bf16 ----------------
__global__ void prep_kernel(const float* __restrict__ W1, const float* __restrict__ W2,
                            const float* g1, const float* b1, const float* m1, const float* v1,
                            const float* g2, const float* b2, const float* m2, const float* v2,
                            bf16* w1bf, bf16* w2bf,
                            float* s1, float* bi1, float* s2, float* bi2)
{
    int g = blockIdx.x * 256 + threadIdx.x;
    const int n1 = 256 * K1;
    if (g < n1) {
        int o = g / K1, c = g % K1;
        w1bf[g] = (c < K1_REAL) ? (bf16)W1[o * K1_REAL + c] : (bf16)0.0f;
    } else if (g < n1 + 256 * 256) {
        int e = g - n1;
        w2bf[e] = (bf16)W2[e];
    } else if (g < n1 + 256 * 256 + 256) {
        int o = g - (n1 + 256 * 256);
        float sa = g1[o] * rsqrtf(v1[o] + 1e-5f);
        s1[o] = sa; bi1[o] = b1[o] - m1[o] * sa;
        float sb = g2[o] * rsqrtf(v2[o] + 1e-5f);
        s2[o] = sb; bi2[o] = b2[o] - m2[o] * sb;
    }
}

// ---------------- select: stable first-64 inside-crop neighbors ----------------
__global__ void __launch_bounds__(256) select_kernel(const float* __restrict__ xyz,
                                                     const float* __restrict__ rot,
                                                     const float* __restrict__ crop,
                                                     int* __restrict__ sel_idx,
                                                     float* __restrict__ sel_loc)
{
    __shared__ float sxyz[NPTS * 3];          // this block's batch xyz
    __shared__ int   idxbuf[8][NS];
    __shared__ float locbuf[8][NS][3];

    int tid  = threadIdx.x;
    int lane = tid & 31, wv = tid >> 5;
    int pt   = blockIdx.x * 8 + wv;           // 0..2047 (b uniform per block)
    int b    = (blockIdx.x * 8) >> 10;
    int p    = pt & (NPTS - 1);

    const float* xb = xyz + (size_t)b * NPTS * 3;
    for (int e = tid; e < NPTS * 3; e += 256) sxyz[e] = xb[e];
    __syncthreads();

    float cx = sxyz[p * 3 + 0], cy = sxyz[p * 3 + 1], cz = sxyz[p * 3 + 2];
    const float* rp = rot + (size_t)pt * 9;   // rot[b,p,j,k]
    float r00 = rp[0], r01 = rp[1], r02 = rp[2];
    float r10 = rp[3], r11 = rp[4], r12 = rp[5];
    float r20 = rp[6], r21 = rp[7], r22 = rp[8];
    const float* cp = crop + (size_t)pt * 3;
    float h0 = 0.5f * cp[0], h1v = 0.5f * cp[1], h2v = 0.5f * cp[2];

    int cnt = 0;
    for (int base = 0; base < NPTS && cnt < NS; base += 32) {
        int n = base + lane;
        float dx = sxyz[n * 3 + 0] - cx;
        float dy = sxyz[n * 3 + 1] - cy;
        float dz = sxyz[n * 3 + 2] - cz;
        float l0 = dx * r00 + dy * r10 + dz * r20;
        float l1 = dx * r01 + dy * r11 + dz * r21;
        float l2 = dx * r02 + dy * r12 + dz * r22;
        bool inside = (fabsf(l0) <= h0) && (fabsf(l1) <= h1v) && (fabsf(l2) <= h2v);
        unsigned m = (unsigned)__ballot(inside);
        int pos = cnt + __popc(m & ((1u << lane) - 1u));
        if (inside && pos < NS) {
            idxbuf[wv][pos] = n;
            locbuf[wv][pos][0] = l0; locbuf[wv][pos][1] = l1; locbuf[wv][pos][2] = l2;
        }
        cnt += __popc(m);
    }
    if (cnt > NS) cnt = NS;
    // pad with slot 0 (same wave: LDS ops are in-order)
    for (int s = cnt + lane; s < NS; s += 32) {
        idxbuf[wv][s] = idxbuf[wv][0];
        locbuf[wv][s][0] = locbuf[wv][0][0];
        locbuf[wv][s][1] = locbuf[wv][0][1];
        locbuf[wv][s][2] = locbuf[wv][0][2];
    }
    for (int s = lane; s < NS; s += 32) {
        sel_idx[pt * NS + s] = idxbuf[wv][s];
        sel_loc[(pt * NS + s) * 3 + 0] = locbuf[wv][s][0];
        sel_loc[(pt * NS + s) * 3 + 1] = locbuf[wv][s][1];
        sel_loc[(pt * NS + s) * 3 + 2] = locbuf[wv][s][2];
    }
}

// ---------------- WMMA fragment loaders (bf16 16x16x32, contiguous k = 16h + s in both A and B) ----------------
// A (16x32, MxK): lane<16 holds row M=lane&15, K in {koff..koff+15},
//                 koff = 0 (lanes 0-15) or 16 (lanes 16-31).
__device__ __forceinline__ v16bf load_a_frag(const bf16* rowbase, int stride, int lane)
{
    int m    = lane & 15;
    int koff = (lane & 16) ? 16 : 0;
    const bf16* r = rowbase + m * stride;
    v8bf lo = *(const v8bf*)(r + koff);
    v8bf hi = *(const v8bf*)(r + 8 + koff);
    v16bf a;
#pragma unroll
    for (int i = 0; i < 8; ++i) { a[i] = lo[i]; a[i + 8] = hi[i]; }
    return a;
}
// B (32x16, KxN): lane&15 selects column N; lanes<16 hold K 0..15, lanes>=16 hold K 16..31.
// Weight stored row-major [n][k] -> 16 contiguous bf16 = one 32B load.
__device__ __forceinline__ v16bf load_b_frag(const bf16* w, int ld, int n0, int kb, int lane)
{
    int n  = n0 + (lane & 15);
    int k2 = kb + ((lane & 16) ? 16 : 0);
    return *(const v16bf*)(w + (size_t)n * ld + k2);
}

// ---------------- fused gather + MLP(2 layers) + max-pool ----------------
// Tile ownership: wave w owns N-tiles {2w, 2w+1} and ALL 4 M-tiles.
//  -> every B (weight) fragment is loaded by exactly one wave (unique L2 traffic)
//  -> the max-over-samples reduction is wave-local (4 mt in-register + shfl_xor(16))
__global__ void __launch_bounds__(256) pointnet_kernel(
    const float* __restrict__ feats,      // (B, C, N)
    const int*   __restrict__ sel_idx,
    const float* __restrict__ sel_loc,
    const bf16*  __restrict__ w1bf,       // 256 x 544 row-major
    const bf16*  __restrict__ w2bf,       // 256 x 256 row-major
    const float* __restrict__ s1, const float* __restrict__ bi1,
    const float* __restrict__ s2, const float* __restrict__ bi2,
    float* __restrict__ out)              // (B, 256, N)
{
    extern __shared__ char smem[];
    bf16* grouped = (bf16*)smem;                   // 64 x GSTRIDE  (70656 B)
    bf16* h1      = (bf16*)(smem + 70656);         // 64 x H1STRIDE (33792 B)

    int tid = threadIdx.x;
    int pt  = blockIdx.x;
    int b   = pt >> 10, p = pt & (NPTS - 1);

    // stage local xyz (cols 0..2) + zero pad cols 515..543
    if (tid < NS) {
        const float* lp = sel_loc + (size_t)(pt * NS + tid) * 3;
        bf16* row = grouped + tid * GSTRIDE;
        row[0] = (bf16)lp[0]; row[1] = (bf16)lp[1]; row[2] = (bf16)lp[2];
#pragma unroll
        for (int j = K1_REAL; j < K1; ++j) row[j] = (bf16)0.0f;
    }
    // gather features: thread owns fixed row m = tid&63, walks channels
    {
        int m = tid & 63;
        int i = sel_idx[pt * NS + m];
        const float* fb = feats + (size_t)b * CFEAT * NPTS + i;
        bf16* row = grouped + m * GSTRIDE + 3;
        for (int c = tid >> 6; c < CFEAT; c += 4)
            row[c] = (bf16)fb[(size_t)c * NPTS];
    }
    __syncthreads();

    int lane = tid & 31, wv = tid >> 5;
    int nt0 = wv * 2;                         // this wave's two N-tiles

    v8f acc[2][4];
#pragma unroll
    for (int j = 0; j < 2; ++j)
#pragma unroll
        for (int mt = 0; mt < 4; ++mt)
#pragma unroll
            for (int r = 0; r < 8; ++r) acc[j][mt][r] = 0.0f;

    // ---- GEMM1: (64x544) x (544x256) ----
    for (int kb = 0; kb < K1; kb += 32) {
        v16bf a0 = load_a_frag(grouped + 0 * 16 * GSTRIDE + kb, GSTRIDE, lane);
        v16bf a1 = load_a_frag(grouped + 1 * 16 * GSTRIDE + kb, GSTRIDE, lane);
        v16bf a2 = load_a_frag(grouped + 2 * 16 * GSTRIDE + kb, GSTRIDE, lane);
        v16bf a3 = load_a_frag(grouped + 3 * 16 * GSTRIDE + kb, GSTRIDE, lane);
#pragma unroll
        for (int j = 0; j < 2; ++j) {
            v16bf bf = load_b_frag(w1bf, K1, (nt0 + j) * 16, kb, lane);
            acc[j][0] = __builtin_amdgcn_wmma_f32_16x16x32_bf16(false, a0, false, bf, (short)0, acc[j][0], false, false);
            acc[j][1] = __builtin_amdgcn_wmma_f32_16x16x32_bf16(false, a1, false, bf, (short)0, acc[j][1], false, false);
            acc[j][2] = __builtin_amdgcn_wmma_f32_16x16x32_bf16(false, a2, false, bf, (short)0, acc[j][2], false, false);
            acc[j][3] = __builtin_amdgcn_wmma_f32_16x16x32_bf16(false, a3, false, bf, (short)0, acc[j][3], false, false);
        }
    }

    // ---- BN1 + ReLU -> h1 (bf16) ----
    int rowoff = (lane & 16) ? 8 : 0;
#pragma unroll
    for (int j = 0; j < 2; ++j) {
        int n = (nt0 + j) * 16 + (lane & 15);
        float sc = s1[n], bb = bi1[n];
#pragma unroll
        for (int mt = 0; mt < 4; ++mt)
#pragma unroll
            for (int r = 0; r < 8; ++r) {
                float y = acc[j][mt][r] * sc + bb;
                y = y > 0.0f ? y : 0.0f;
                h1[(mt * 16 + rowoff + r) * H1STRIDE + n] = (bf16)y;
            }
    }
    __syncthreads();

    // ---- GEMM2: (64x256) x (256x256) ----
#pragma unroll
    for (int j = 0; j < 2; ++j)
#pragma unroll
        for (int mt = 0; mt < 4; ++mt)
#pragma unroll
            for (int r = 0; r < 8; ++r) acc[j][mt][r] = 0.0f;

    for (int kb = 0; kb < 256; kb += 32) {
        v16bf a0 = load_a_frag(h1 + 0 * 16 * H1STRIDE + kb, H1STRIDE, lane);
        v16bf a1 = load_a_frag(h1 + 1 * 16 * H1STRIDE + kb, H1STRIDE, lane);
        v16bf a2 = load_a_frag(h1 + 2 * 16 * H1STRIDE + kb, H1STRIDE, lane);
        v16bf a3 = load_a_frag(h1 + 3 * 16 * H1STRIDE + kb, H1STRIDE, lane);
#pragma unroll
        for (int j = 0; j < 2; ++j) {
            v16bf bf = load_b_frag(w2bf, 256, (nt0 + j) * 16, kb, lane);
            acc[j][0] = __builtin_amdgcn_wmma_f32_16x16x32_bf16(false, a0, false, bf, (short)0, acc[j][0], false, false);
            acc[j][1] = __builtin_amdgcn_wmma_f32_16x16x32_bf16(false, a1, false, bf, (short)0, acc[j][1], false, false);
            acc[j][2] = __builtin_amdgcn_wmma_f32_16x16x32_bf16(false, a2, false, bf, (short)0, acc[j][2], false, false);
            acc[j][3] = __builtin_amdgcn_wmma_f32_16x16x32_bf16(false, a3, false, bf, (short)0, acc[j][3], false, false);
        }
    }

    // ---- BN2 + ReLU + max over all 64 samples (wave-local) ----
#pragma unroll
    for (int j = 0; j < 2; ++j) {
        int n = (nt0 + j) * 16 + (lane & 15);
        float sc = s2[n], bb = bi2[n];
        float v = 0.0f;                       // post-ReLU values are >= 0
#pragma unroll
        for (int mt = 0; mt < 4; ++mt)
#pragma unroll
            for (int r = 0; r < 8; ++r) {
                float y = acc[j][mt][r] * sc + bb;
                y = y > 0.0f ? y : 0.0f;
                v = fmaxf(v, y);
            }
        v = fmaxf(v, __shfl_xor(v, 16, 32));  // combine the two row halves
        if (lane < 16)
            out[((size_t)b * 256 + n) * NPTS + p] = v;
    }
}

extern "C" void kernel_launch(void* const* d_in, const int* in_sizes, int n_in,
                              void* d_out, int out_size, void* d_ws, size_t ws_size,
                              hipStream_t stream)
{
    const float* xyz   = (const float*)d_in[0];
    const float* feats = (const float*)d_in[1];
    const float* rot   = (const float*)d_in[2];
    const float* crop  = (const float*)d_in[3];
    const float* W1    = (const float*)d_in[4];
    const float* g1    = (const float*)d_in[5];
    const float* b1    = (const float*)d_in[6];
    const float* m1    = (const float*)d_in[7];
    const float* v1    = (const float*)d_in[8];
    const float* W2    = (const float*)d_in[9];
    const float* g2    = (const float*)d_in[10];
    const float* b2    = (const float*)d_in[11];
    const float* m2    = (const float*)d_in[12];
    const float* v2    = (const float*)d_in[13];

    char* ws = (char*)d_ws;
    bf16*  w1bf    = (bf16*)(ws + WS_W1BF);
    bf16*  w2bf    = (bf16*)(ws + WS_W2BF);
    float* s1      = (float*)(ws + WS_SCALE1);
    float* bi1     = (float*)(ws + WS_BIAS1);
    float* s2      = (float*)(ws + WS_SCALE2);
    float* bi2     = (float*)(ws + WS_BIAS2);
    int*   sel_idx = (int*)(ws + WS_SELIDX);
    float* sel_loc = (float*)(ws + WS_SELLOC);

    // prep: 256*544 + 256*256 + 256 = 205056 elements -> 801 blocks
    prep_kernel<<<801, 256, 0, stream>>>(W1, W2, g1, b1, m1, v1, g2, b2, m2, v2,
                                         w1bf, w2bf, s1, bi1, s2, bi2);
    // select: 2048 points, 8 per block
    select_kernel<<<256, 256, 0, stream>>>(xyz, rot, crop, sel_idx, sel_loc);

    (void)hipFuncSetAttribute((const void*)pointnet_kernel,
                              hipFuncAttributeMaxDynamicSharedMemorySize, 104448);
    pointnet_kernel<<<2048, 256, 104448, stream>>>(feats, sel_idx, sel_loc,
                                                   w1bf, w2bf, s1, bi1, s2, bi2,
                                                   (float*)d_out);
}
